// CapsuleLayer_48455821033882
// MI455X (gfx1250) — matrix operand order repaired and checked by reading, not compiled, not run on hardware
//
#include <hip/hip_runtime.h>
#include <hip/hip_bf16.h>

typedef __attribute__((ext_vector_type(16))) __bf16 v16bf;
typedef __attribute__((ext_vector_type(8)))  float  v8f;

#define NN   2048   // n_nodes
#define INC  16     // in_C
#define NC   32     // n_caps
#define OC   32     // out_C
#define KO   1024   // NC*OC
#define BB   64     // batch

// ---------------------------------------------------------------------------
// Kernel 1: priors[b,n,k,o] = sum_i x[b,n,i] * W[n,k,i,o], stored bf16.
// One block per node n; 8 waves; wave computes 32 of the 4x64 16x16 tiles of
// the 64x1024 per-node GEMM using v_wmma_f32_16x16x32_bf16 (K padded 16->32).
// ---------------------------------------------------------------------------
__global__ __launch_bounds__(256) void gemm_priors(
    const float* __restrict__ x,   // [64,2048,16]
    const float* __restrict__ w,   // [2048,32,16,32]
    __bf16* __restrict__ P)        // [64,2048,1024]
{
  const int n      = blockIdx.x;
  const int lane   = threadIdx.x & 31;
  const int wv     = threadIdx.x >> 5;   // 0..7
  const int mt     = wv & 3;             // M tile (b dim)
  const int ntBase = wv >> 2;            // 0 or 1 (tile parity)
  const int half   = lane >> 4;
  const int r16    = lane & 15;
  const int b0     = mt * 16;

  // A fragment: 16x32 bf16 (MxK). lanes 0-15: elems0-7 = K0..7, elems8-15 = K16..23(zero)
  //             lanes16-31: elems0-7 = K8..15, elems8-15 = K24..31(zero)
  const int kbase = half * 8;
  const float* xrow = x + ((size_t)(b0 + r16) * NN + n) * INC + kbase;
  float4 xa = *(const float4*)(xrow);
  float4 xb = *(const float4*)(xrow + 4);
  v16bf A;
  A[0] = (__bf16)xa.x; A[1] = (__bf16)xa.y; A[2] = (__bf16)xa.z; A[3] = (__bf16)xa.w;
  A[4] = (__bf16)xb.x; A[5] = (__bf16)xb.y; A[6] = (__bf16)xb.z; A[7] = (__bf16)xb.w;
#pragma unroll
  for (int e = 8; e < 16; ++e) A[e] = (__bf16)0.0f;

  const float* wbase = w + (size_t)n * (NC * INC * OC);

  for (int j = 0; j < 32; ++j) {
    const int nt   = ntBase + 2 * j;
    const int col  = nt * 16 + r16;        // 0..1023
    const int kcap = col >> 5;
    const int o    = col & 31;

    // B fragment: 32x16 bf16 (KxN). lanes 0-15: elems0-7 = K0..7, elems8-15 = K16..23(zero)
    //             lanes16-31: elems0-7 = K8..15, elems8-15 = K24..31(zero)
    v16bf Bv;
    {
      const float* bp = wbase + kcap * (INC * OC) + o + half * 8 * OC;   // stride OC between i
#pragma unroll
      for (int e = 0; e < 8; ++e) Bv[e] = (__bf16)bp[e * OC];
#pragma unroll
      for (int e = 8; e < 16; ++e) Bv[e] = (__bf16)0.0f;
    }

    v8f C = {0.f, 0.f, 0.f, 0.f, 0.f, 0.f, 0.f, 0.f};
    C = __builtin_amdgcn_wmma_f32_16x16x32_bf16(false, A, false, Bv,
                                                (short)0, C, false, false);

    // D layout: reg r -> (M = r + half*8, N = r16)
#pragma unroll
    for (int r = 0; r < 8; ++r) {
      size_t idx = ((size_t)(b0 + r + half * 8) * NN + n) * KO + col;
      P[idx] = (__bf16)C[r];
    }
  }
}

// ---------------------------------------------------------------------------
// Kernel 2: s0 partials (uniform probs 1/32): parts[b,c,ko] = (1/32)*sum_{n in c} P
// Grid: 64 b * 64 chunks (32 nodes each). Thread handles 4 consecutive ko.
// ---------------------------------------------------------------------------
__global__ __launch_bounds__(256) void s0_reduce(
    const __bf16* __restrict__ P, float* __restrict__ parts)
{
  const int b = blockIdx.x >> 6;
  const int c = blockIdx.x & 63;
  const int t = threadIdx.x;
  const __bf16* base = P + ((size_t)b * NN + (size_t)c * 32) * KO + t * 4;
  float a0 = 0.f, a1 = 0.f, a2 = 0.f, a3 = 0.f;
  for (int n = 0; n < 32; ++n) {
    uint2 u = *(const uint2*)(base + (size_t)n * KO);
    a0 += __uint_as_float(u.x << 16);
    a1 += __uint_as_float(u.x & 0xFFFF0000u);
    a2 += __uint_as_float(u.y << 16);
    a3 += __uint_as_float(u.y & 0xFFFF0000u);
  }
  const float inv = 1.0f / 32.0f;
  float4 outv = {a0 * inv, a1 * inv, a2 * inv, a3 * inv};
  *(float4*)(parts + (size_t)(b * 64 + c) * KO + t * 4) = outv;
}

// ---------------------------------------------------------------------------
// Kernel 3: reduce chunk partials -> s[b,k,o], squash over o, write V (and out).
// One block per b.
// ---------------------------------------------------------------------------
__global__ __launch_bounds__(256) void squash_k(
    const float* __restrict__ parts, float* __restrict__ V, float* __restrict__ out)
{
  __shared__ float s[KO];
  __shared__ float scale[NC];
  const int b = blockIdx.x;
  const int t = threadIdx.x;
  const float* pb = parts + (size_t)b * 64 * KO;
#pragma unroll
  for (int j = 0; j < 4; ++j) {
    int p = t + 256 * j;
    float acc = 0.f;
    for (int c = 0; c < 64; ++c) acc += pb[(size_t)c * KO + p];
    s[p] = acc;
  }
  __syncthreads();
  if (t < NC) {
    float ss = 0.f;
#pragma unroll
    for (int o = 0; o < OC; ++o) { float v = s[t * OC + o]; ss += v * v; }
    scale[t] = (ss / (1.0f + ss)) * rsqrtf(ss);
  }
  __syncthreads();
#pragma unroll
  for (int j = 0; j < 4; ++j) {
    int p = t + 256 * j;
    float v = s[p] * scale[p >> 5];
    V[(size_t)b * KO + p] = v;
    if (out) out[(size_t)b * KO + p] = v;
  }
}

// ---------------------------------------------------------------------------
// Kernel 4: fused logits-update + softmax + next-iteration s partials.
// lane = capsule k. delta is a per-lane dot; softmax over caps = over lanes
// (2 wave butterflies). s accumulated per-lane, combined via padded LDS.
// ---------------------------------------------------------------------------
__global__ __launch_bounds__(256) void route_update(
    const __bf16* __restrict__ P, const float* __restrict__ V,
    float* __restrict__ L, float* __restrict__ parts,
    int addL, int storeL)
{
  const int b    = blockIdx.x >> 6;
  const int c    = blockIdx.x & 63;
  const int lane = threadIdx.x & 31;   // capsule k
  const int wv   = threadIdx.x >> 5;   // 0..7

  float Vr[OC];
  const float4* vb4 = (const float4*)(V + ((size_t)b * NC + lane) * OC);
#pragma unroll
  for (int q = 0; q < 8; ++q) {
    float4 tv = vb4[q];
    Vr[4 * q + 0] = tv.x; Vr[4 * q + 1] = tv.y;
    Vr[4 * q + 2] = tv.z; Vr[4 * q + 3] = tv.w;
  }
  float sacc[OC];
#pragma unroll
  for (int o = 0; o < OC; ++o) sacc[o] = 0.f;

  for (int j = 0; j < 4; ++j) {
    const int n = c * 32 + wv * 4 + j;
    const __bf16* pr = P + ((size_t)b * NN + n) * KO + lane * OC;
    float Pf[OC];
#pragma unroll
    for (int q = 0; q < 4; ++q) {
      uint4 u = *(const uint4*)(pr + q * 8);
      Pf[q * 8 + 0] = __uint_as_float(u.x << 16);
      Pf[q * 8 + 1] = __uint_as_float(u.x & 0xFFFF0000u);
      Pf[q * 8 + 2] = __uint_as_float(u.y << 16);
      Pf[q * 8 + 3] = __uint_as_float(u.y & 0xFFFF0000u);
      Pf[q * 8 + 4] = __uint_as_float(u.z << 16);
      Pf[q * 8 + 5] = __uint_as_float(u.z & 0xFFFF0000u);
      Pf[q * 8 + 6] = __uint_as_float(u.w << 16);
      Pf[q * 8 + 7] = __uint_as_float(u.w & 0xFFFF0000u);
    }
    float delta = 0.f;
#pragma unroll
    for (int o = 0; o < OC; ++o) delta = fmaf(Pf[o], Vr[o], delta);

    float l = delta;
    const size_t lidx = ((size_t)b * NN + n) * NC + lane;
    if (addL)   l += L[lidx];
    if (storeL) L[lidx] = l;

    // softmax over lanes (caps)
    float m = l;
    for (int d = 1; d < 32; d <<= 1) m = fmaxf(m, __shfl_xor(m, d));
    float e = __expf(l - m);
    float ssum = e;
    for (int d = 1; d < 32; d <<= 1) ssum += __shfl_xor(ssum, d);
    float p = __fdividef(e, ssum);

#pragma unroll
    for (int o = 0; o < OC; ++o) sacc[o] = fmaf(p, Pf[o], sacc[o]);
  }

  // combine 8 waves via padded LDS (stride 33 -> conflict-free)
  __shared__ float ls[8 * 1056];
  float* mine = ls + wv * 1056 + lane * 33;
#pragma unroll
  for (int o = 0; o < OC; ++o) mine[o] = sacc[o];
  __syncthreads();

  const int t = threadIdx.x;
  float* pout = parts + (size_t)(b * 64 + c) * KO;
#pragma unroll
  for (int j2 = 0; j2 < 4; ++j2) {
    int p2 = t + 256 * j2;
    int k2 = p2 >> 5, o2 = p2 & 31;
    float acc = 0.f;
#pragma unroll
    for (int w2 = 0; w2 < 8; ++w2) acc += ls[w2 * 1056 + k2 * 33 + o2];
    pout[p2] = acc;
  }
}

// ---------------------------------------------------------------------------
extern "C" void kernel_launch(void* const* d_in, const int* in_sizes, int n_in,
                              void* d_out, int out_size, void* d_ws, size_t ws_size,
                              hipStream_t stream) {
  const float* x = (const float*)d_in[0];
  const float* w = (const float*)d_in[1];
  float* out = (float*)d_out;

  char* ws = (char*)d_ws;
  __bf16* P    = (__bf16*)ws;                                   // 268,435,456 B
  float* L     = (float*)(ws + 268435456ull);                   //  16,777,216 B
  float* parts = (float*)(ws + 268435456ull + 16777216ull);     //  16,777,216 B
  float* V     = (float*)(ws + 268435456ull + 2ull * 16777216); //     262,144 B

  gemm_priors <<<NN,   256, 0, stream>>>(x, w, P);
  s0_reduce   <<<BB*64,256, 0, stream>>>(P, parts);
  squash_k    <<<BB,   256, 0, stream>>>(parts, V, nullptr);           // V0
  route_update<<<BB*64,256, 0, stream>>>(P, V, L, parts, 0, 1);        // L=delta0, s1
  squash_k    <<<BB,   256, 0, stream>>>(parts, V, nullptr);           // V1
  route_update<<<BB*64,256, 0, stream>>>(P, V, L, parts, 1, 0);        // s2
  squash_k    <<<BB,   256, 0, stream>>>(parts, V, out);               // V2 -> out
}
